// NeuralCDE_32993938768495
// MI455X (gfx1250) — compile-verified, matrix operand order checked by reading, not run on hardware
//
#include <hip/hip_runtime.h>
#include <hip/hip_bf16.h>

typedef __bf16 bf16;
typedef __attribute__((ext_vector_type(16))) __bf16 v16bf;
typedef __attribute__((ext_vector_type(8)))  float  v8f;
typedef __attribute__((ext_vector_type(4)))  float  f32x4;

#define B_DIM 512
#define H_DIM 128
#define D_DIM 16
#define W_DIM 512
#define N_KNOT 201
#define DT_F 0.01f
#define STEPS 200
#define F_LOR 8.0f

// 32-byte WMMA operand fragment: two contiguous 16B chunks per lane.
union Frag { v16bf v; f32x4 q[2]; };

__device__ __forceinline__ float softplus_f(float x) {
    return (x > 20.0f) ? x : log1pf(__expf(x));
}

// CDNA5 16-bit A/B fragment layout (ISA 7.12.2): lane l holds row (l&15),
// K elements (l>>4)*8 + {0..7} in halves 0..7 and + {16..23} in halves 8..15.
// Source is row-major [rows, ld] bf16 with K contiguous -> two b128 loads.
__device__ __forceinline__ void load_frag(Frag& f, const bf16* __restrict__ base,
                                          int row, int ld, int k0, int lane) {
    const bf16* p = base + (size_t)row * ld + k0 + ((lane >> 4) << 3);
    f.q[0] = *reinterpret_cast<const f32x4*>(p);
    f.q[1] = *reinterpret_cast<const f32x4*>(p + 16);
}

// C = softplus(A[M,K] @ W[N,K]^T + bias), output stored bf16 row-major [M,N].
// One wave computes a 32x64 block: 2 M-tiles x 4 N-tiles of 16x16.
__global__ __launch_bounds__(256) void gemm_bf16_softplus(
    const bf16* __restrict__ A, const bf16* __restrict__ Wt,
    const float* __restrict__ bias, bf16* __restrict__ Out,
    int M, int N, int K) {
    const int lane = threadIdx.x & 31;
    const int wid  = blockIdx.x * (blockDim.x >> 5) + (threadIdx.x >> 5);
    const int mtiles = M >> 5;
    const int m0 = (wid % mtiles) << 5;
    const int n0 = (wid / mtiles) << 6;
    const int r16 = lane & 15;
    const int hi  = lane >> 4;

    v8f c[2][4] = {};
    for (int k0 = 0; k0 < K; k0 += 32) {
        Frag a0, a1, b;
        load_frag(a0, A, m0 + r16,      K, k0, lane);
        load_frag(a1, A, m0 + 16 + r16, K, k0, lane);
#pragma unroll
        for (int nt = 0; nt < 4; ++nt) {
            load_frag(b, Wt, n0 + nt * 16 + r16, K, k0, lane);
            c[0][nt] = __builtin_amdgcn_wmma_f32_16x16x32_bf16(
                false, a0.v, false, b.v, (short)0, c[0][nt], false, false);
            c[1][nt] = __builtin_amdgcn_wmma_f32_16x16x32_bf16(
                false, a1.v, false, b.v, (short)0, c[1][nt], false, false);
        }
    }
#pragma unroll
    for (int mt = 0; mt < 2; ++mt)
#pragma unroll
        for (int nt = 0; nt < 4; ++nt) {
            const int n = n0 + nt * 16 + r16;
            const float bv = bias[n];
#pragma unroll
            for (int r = 0; r < 8; ++r) {
                const int m = m0 + mt * 16 + r + (hi << 3);
                Out[(size_t)m * N + n] = (bf16)softplus_f(c[mt][nt][r] + bv);
            }
        }
}

// G = tanh(A @ W2^T + b2) then R[b,h] = sum_d G[b, h*16+d] * dX[b,d].
// An aligned 16-wide N-tile covers exactly one h (d = lane&15), so the
// contraction is a 16-lane shuffle reduction inside the C fragment.
__global__ __launch_bounds__(256) void gemm_bf16_tanh_dx(
    const bf16* __restrict__ A, const bf16* __restrict__ Wt,
    const float* __restrict__ bias, const float* __restrict__ dX,
    float* __restrict__ R, int M, int N, int K) {
    const int lane = threadIdx.x & 31;
    const int wid  = blockIdx.x * (blockDim.x >> 5) + (threadIdx.x >> 5);
    const int mtiles = M >> 5;
    const int m0 = (wid % mtiles) << 5;
    const int n0 = (wid / mtiles) << 6;
    const int r16 = lane & 15;
    const int hi  = lane >> 4;

    v8f c[2][4] = {};
    for (int k0 = 0; k0 < K; k0 += 32) {
        Frag a0, a1, b;
        load_frag(a0, A, m0 + r16,      K, k0, lane);
        load_frag(a1, A, m0 + 16 + r16, K, k0, lane);
#pragma unroll
        for (int nt = 0; nt < 4; ++nt) {
            load_frag(b, Wt, n0 + nt * 16 + r16, K, k0, lane);
            c[0][nt] = __builtin_amdgcn_wmma_f32_16x16x32_bf16(
                false, a0.v, false, b.v, (short)0, c[0][nt], false, false);
            c[1][nt] = __builtin_amdgcn_wmma_f32_16x16x32_bf16(
                false, a1.v, false, b.v, (short)0, c[1][nt], false, false);
        }
    }
#pragma unroll
    for (int mt = 0; mt < 2; ++mt)
#pragma unroll
        for (int nt = 0; nt < 4; ++nt) {
            const int h = (n0 + nt * 16) >> 4;     // one h per 16-wide N-tile
            const int d = r16;
            const float bv = bias[n0 + nt * 16 + d];
#pragma unroll
            for (int r = 0; r < 8; ++r) {
                const int m = m0 + mt * 16 + r + (hi << 3);
                float p = tanhf(c[mt][nt][r] + bv) * dX[m * D_DIM + d];
                p += __shfl_xor(p, 1);
                p += __shfl_xor(p, 2);
                p += __shfl_xor(p, 4);
                p += __shfl_xor(p, 8);
                if (r16 == 0) R[(size_t)m * H_DIM + h] = p;
            }
        }
}

// dX[b,d] = interp(t1) - interp(t0), diffrax cubic: a + f*(b + f*(c + f*d)).
__global__ void cde_dx(const float* __restrict__ ca, const float* __restrict__ cb,
                       const float* __restrict__ cc, const float* __restrict__ cd,
                       float* __restrict__ dX, int step) {
    int i = blockIdx.x * blockDim.x + threadIdx.x;
    if (i >= B_DIM * D_DIM) return;
    const int b = i >> 4, d = i & 15;
    const float t0 = step * DT_F, t1 = t0 + DT_F;
    auto ev = [&](float t) {
        int idx = (int)floorf(t * 100.0f + 1e-3f);     // ts[i] = i * 0.01
        idx = idx < 0 ? 0 : (idx > N_KNOT - 2 ? N_KNOT - 2 : idx);
        const float f = t - idx * DT_F;
        const size_t base = ((size_t)b * (N_KNOT - 1) + idx) * D_DIM + d;
        return ca[base] + f * (cb[base] + f * (cc[base] + f * cd[base]));
    };
    dX[i] = ev(t1) - ev(t0);
}

// u_new = u + lorenz96(u)*DT + R ; also emit bf16 copy for next step's GEMM A.
__global__ void cde_update(const float* __restrict__ u, const float* __restrict__ R,
                           float* __restrict__ u_out, bf16* __restrict__ u_bf) {
    int i = blockIdx.x * blockDim.x + threadIdx.x;
    if (i >= B_DIM * H_DIM) return;
    const int b = i >> 7, h = i & 127;
    const float* ub = u + ((size_t)b << 7);
    const float x   = ub[h];
    const float xp1 = ub[(h + 1) & 127];
    const float xm1 = ub[(h + 127) & 127];
    const float xm2 = ub[(h + 126) & 127];
    const float un = x + ((xp1 - xm2) * xm1 - x + F_LOR) * DT_F + R[i];
    u_out[i] = un;
    u_bf[i]  = (bf16)un;
}

__global__ void cvt_f32_bf16(const float* __restrict__ src, bf16* __restrict__ dst, int n) {
    int i = blockIdx.x * blockDim.x + threadIdx.x;
    if (i < n) dst[i] = (bf16)src[i];
}

extern "C" void kernel_launch(void* const* d_in, const int* in_sizes, int n_in,
                              void* d_out, int out_size, void* d_ws, size_t ws_size,
                              hipStream_t stream) {
    (void)in_sizes; (void)n_in; (void)out_size; (void)ws_size;
    const float* u0 = (const float*)d_in[0];
    const float* ca = (const float*)d_in[2];
    const float* cb = (const float*)d_in[3];
    const float* cc = (const float*)d_in[4];
    const float* cd = (const float*)d_in[5];
    const float* W0 = (const float*)d_in[6];
    const float* b0 = (const float*)d_in[7];
    const float* W1 = (const float*)d_in[8];
    const float* b1 = (const float*)d_in[9];
    const float* W2 = (const float*)d_in[10];
    const float* b2 = (const float*)d_in[11];

    char* ws = (char*)d_ws;
    size_t off = 0;
    auto alloc = [&](size_t bytes) -> void* {
        void* p = ws + off;
        off += (bytes + 255) & ~(size_t)255;
        return p;
    };
    bf16* W0b  = (bf16*)alloc((size_t)W_DIM * H_DIM * 2);          // [512,128]
    bf16* W1b  = (bf16*)alloc((size_t)W_DIM * W_DIM * 2);          // [512,512]
    bf16* W2b  = (bf16*)alloc((size_t)H_DIM * D_DIM * W_DIM * 2);  // [2048,512]
    bf16* ubf  = (bf16*)alloc((size_t)B_DIM * H_DIM * 2);
    float* up0 = (float*)alloc((size_t)B_DIM * H_DIM * 4);
    float* up1 = (float*)alloc((size_t)B_DIM * H_DIM * 4);
    float* dXb = (float*)alloc((size_t)B_DIM * D_DIM * 4);
    float* Rb  = (float*)alloc((size_t)B_DIM * H_DIM * 4);
    bf16* H1   = (bf16*)alloc((size_t)B_DIM * W_DIM * 2);
    bf16* H2   = (bf16*)alloc((size_t)B_DIM * W_DIM * 2);
    float* upp[2] = {up0, up1};

    // One-time (per capture) bf16 staging of weights + initial state.
    cvt_f32_bf16<<<(W_DIM * H_DIM + 255) / 256, 256, 0, stream>>>(W0, W0b, W_DIM * H_DIM);
    cvt_f32_bf16<<<(W_DIM * W_DIM + 255) / 256, 256, 0, stream>>>(W1, W1b, W_DIM * W_DIM);
    cvt_f32_bf16<<<(H_DIM * D_DIM * W_DIM + 255) / 256, 256, 0, stream>>>(W2, W2b, H_DIM * D_DIM * W_DIM);
    cvt_f32_bf16<<<(B_DIM * H_DIM + 255) / 256, 256, 0, stream>>>(u0, ubf, B_DIM * H_DIM);

    // waves = (M/32)*(N/64); 8 waves (256 threads) per block
    const int blocks_g1 = (B_DIM / 32) * (W_DIM / 64) / 8;            // 16
    const int blocks_g3 = (B_DIM / 32) * ((H_DIM * D_DIM) / 64) / 8;  // 64

    for (int n = 0; n < STEPS; ++n) {
        cde_dx<<<(B_DIM * D_DIM + 255) / 256, 256, 0, stream>>>(ca, cb, cc, cd, dXb, n);
        gemm_bf16_softplus<<<blocks_g1, 256, 0, stream>>>(ubf, W0b, b0, H1,
                                                          B_DIM, W_DIM, H_DIM);
        gemm_bf16_softplus<<<blocks_g1, 256, 0, stream>>>(H1, W1b, b1, H2,
                                                          B_DIM, W_DIM, W_DIM);
        gemm_bf16_tanh_dx<<<blocks_g3, 256, 0, stream>>>(H2, W2b, b2, dXb, Rb,
                                                         B_DIM, H_DIM * D_DIM, W_DIM);
        const float* ucur = (n == 0) ? u0 : upp[n & 1];
        float* unext = (n == STEPS - 1) ? (float*)d_out : upp[(n + 1) & 1];
        cde_update<<<(B_DIM * H_DIM + 255) / 256, 256, 0, stream>>>(ucur, Rb, unext, ubf);
    }
}
